// EvoLSTMCell_59957743452547
// MI455X (gfx1250) — matrix operand order repaired and checked by reading, compiled, not run on hardware
//
#include <hip/hip_runtime.h>
#include <hip/hip_bf16.h>

typedef __attribute__((ext_vector_type(16))) _Float16 v16h;
typedef __attribute__((ext_vector_type(8)))  _Float16 v8h;
typedef __attribute__((ext_vector_type(8)))  float    v8f;

#define NB 32
#define NN 4096
#define NF 128
#define BN (NB*NN)            // 131072 rows

// ---- workspace layout (bytes) ----
#define WS_SUMS 0             // f32[64]  (Ssum[32], Rsum[32])
#define WS_COLS 256           // f32[9*128]
#define WS_CURA 4864          // f32[32]
#define WS_PACK 5120          // f16[5*16384]
#define WS_PART 168960        // f32[4096*128]
#define WS_H    2266112       // f16[BN*128]
// pack sub-offsets in halves
#define PK_WI 0
#define PK_WF 16384
#define PK_WO 32768
#define PK_WC 49152
#define PK_UC 65536

__device__ __forceinline__ float fast_sigmoid(float x) {
    return __builtin_amdgcn_rcpf(1.0f + __expf(-x));
}
__device__ __forceinline__ float fast_tanh(float x) {
    // exact at saturation: exp(+inf)->1, exp(-inf)->-1
    return 1.0f - 2.0f * __builtin_amdgcn_rcpf(__expf(2.0f * x) + 1.0f);
}
__device__ __forceinline__ v8f wmma16(v16h a, v16h b, v8f c) {
    return __builtin_amdgcn_wmma_f32_16x16x32_f16(false, a, false, b, (short)0, c, false, false);
}

// ---------------- prep: row sums of send/receive ----------------
__global__ void k_rowsum(const float* __restrict__ send, const float* __restrict__ recv,
                         float* __restrict__ sums) {
    int mb = blockIdx.x, which = mb >> 5, b = mb & 31;   // grid 64, 256 thr
    const float* src = which ? recv : send;
    __shared__ float red[256];
    float s = 0.f;
    for (int j = threadIdx.x; j < NN; j += 256) s += src[b * NN + j];
    red[threadIdx.x] = s; __syncthreads();
    for (int st = 128; st > 0; st >>= 1) {
        if ((int)threadIdx.x < st) red[threadIdx.x] += red[threadIdx.x + st];
        __syncthreads();
    }
    if (threadIdx.x == 0) sums[which * 32 + b] = red[0];
}

// ---------------- prep: all column sums ----------------
__global__ void k_colsum(const float* Win, const float* Wout,
                         const float* Wi, const float* Wf, const float* Wo, const float* Wc,
                         const float* Ui, const float* Uf, const float* Uo,
                         float* __restrict__ cols) {
    int t = blockIdx.x, f = threadIdx.x;      // grid 9, 128 threads
    const float* src;
    switch (t) {
        case 0: src = Win;  break;
        case 1: src = Wout; break;
        case 2: src = Wi;   break;   // top 128 rows of (256,128)
        case 3: src = Wf;   break;
        case 4: src = Wo;   break;
        case 5: src = Wc;   break;
        case 6: src = Ui;   break;
        case 7: src = Uf;   break;
        default: src = Uo;  break;
    }
    float s = 0.f;
    for (int k = 0; k < 128; ++k) s += src[k * 128 + f];
    cols[t * 128 + f] = s;
}

// ---------------- prep: pack f16 weights into WMMA B-fragment order ----------------
// B frag (16x16x32 f16): lane holds column n = lane&15, k = kt*32 + 16*(lane>>4) + e
__global__ void k_pack(const float* Wi, const float* Wf, const float* Wo, const float* Wc,
                       const float* Uc, _Float16* __restrict__ pack) {
    int gid = blockIdx.x * 256 + threadIdx.x;   // 5*16384 = 81920 threads
    int m = gid >> 14;
    int idx = gid & 16383;
    int tile = idx >> 9;          // kt*8+ft
    int within = idx & 511;
    int lane = within >> 4;
    int e = within & 15;
    int kt = tile >> 3, ft = tile & 7;
    int k = kt * 32 + ((lane >> 4) << 4) + e;
    int f = ft * 16 + (lane & 15);
    const float* src; int rowoff;
    switch (m) {
        case 0: src = Wi; rowoff = 128; break;
        case 1: src = Wf; rowoff = 128; break;
        case 2: src = Wo; rowoff = 128; break;
        case 3: src = Wc; rowoff = 128; break;
        default: src = Uc; rowoff = 0;  break;
    }
    pack[gid] = (_Float16)src[(size_t)(k + rowoff) * 128 + f];
}

// ---------------- H embedding (elementwise) + deterministic partial col-sums ----------------
__global__ void k_embed(const float* __restrict__ send, const float* __restrict__ recv,
                        const float* __restrict__ b_in, const float* __restrict__ b_out,
                        const float* __restrict__ sums, const float* __restrict__ cols,
                        _Float16* __restrict__ H, float* __restrict__ part) {
    int blk = blockIdx.x;              // 4096 blocks, 32 rows each
    int b = blk >> 7;
    int r0 = (blk & 127) * 32;
    int f = threadIdx.x & 127;
    int rr = threadIdx.x >> 7;         // 0/1
    float c0, c1, d0, d1, msum; const float* node;
    if (f < 64) {                      // pair of Hin columns
        c0 = cols[2 * f];        c1 = cols[2 * f + 1];       // WinCS
        d0 = b_in[2 * f];        d1 = b_in[2 * f + 1];
        node = send + (size_t)b * NN;  msum = sums[32 + b];  // Rsum
    } else {
        int fp = f - 64;
        c0 = cols[128 + 2 * fp]; c1 = cols[128 + 2 * fp + 1]; // WoutCS
        d0 = b_out[2 * fp];      d1 = b_out[2 * fp + 1];
        node = recv + (size_t)b * NN;  msum = sums[b];        // Ssum
    }
    float acc = 0.f;
    for (int r = rr; r < 32; r += 2) {
        float s = node[r0 + r] * msum;
        float h = fmaxf(fast_tanh(s * c0 + d0), fast_tanh(s * c1 + d1));
        H[(size_t)(b * NN + r0 + r) * 128 + f] = (_Float16)h;
        acc += h;
    }
    __shared__ float ps[2][128];
    ps[rr][f] = acc; __syncthreads();
    if (rr == 0) part[(size_t)blk * 128 + f] = ps[0][f] + ps[1][f];
}

// ---------------- cur_a = mean_n(H) @ Wa ----------------
__global__ void k_cura(const float* __restrict__ part, const float* __restrict__ Wa,
                       float* __restrict__ cura, float* __restrict__ out_tail) {
    int b = blockIdx.x, f = threadIdx.x;   // 32 blocks, 128 threads
    float s = 0.f;
    for (int blk = 0; blk < 128; ++blk) s += part[(size_t)(b * 128 + blk) * 128 + f];
    s *= Wa[f];
    __shared__ float red[128];
    red[f] = s; __syncthreads();
    for (int st = 64; st > 0; st >>= 1) {
        if (f < st) red[f] += red[f + st];
        __syncthreads();
    }
    if (f == 0) { float v = red[0] * (1.0f / (float)NN); cura[b] = v; out_tail[b] = v; }
}

// A frag (16x16x32 f16): lane r=lane&15 holds row m=r; k = base + 16*half + e
__device__ __forceinline__ v16h load_a(const _Float16* base, int r, int half, int kt) {
    const _Float16* p = base + r * 136 + kt * 32 + 16 * half;
    v8h lo = *(const v8h*)p;
    v8h hi = *(const v8h*)(p + 8);
    v16h a;
#pragma unroll
    for (int i = 0; i < 8; ++i) { a[i] = lo[i]; a[i + 8] = hi[i]; }
    return a;
}

// ---------------- fused gate GEMMs + epilogue ----------------
__global__ __launch_bounds__(32)
void k_gates(const _Float16* __restrict__ H, const _Float16* __restrict__ pack,
             const float* __restrict__ cols, const float* __restrict__ cura,
             const float* __restrict__ bf, const float* __restrict__ bo,
             const float* __restrict__ bc, float* __restrict__ out) {
    __shared__ _Float16 Hl[16 * 136];
    __shared__ _Float16 Gl[16 * 136];
    const int lane = threadIdx.x;
    const int r = lane & 15, half = lane >> 4;
    const int t0 = blockIdx.x * 16;          // global row base
    const int b = t0 >> 12;

    // stage H tile (16 rows x 128 f16) into LDS: batch loads, one wait, batch stores
    {
        int row = lane >> 1;
        int c0 = (lane & 1) * 64;
        const float4* src = (const float4*)(H + (size_t)(t0 + row) * 128 + c0);
        float4* dst = (float4*)(Hl + row * 136 + c0);
        float4 tmp[8];
#pragma unroll
        for (int j = 0; j < 8; ++j) tmp[j] = src[j];
#pragma unroll
        for (int j = 0; j < 8; ++j) dst[j] = tmp[j];
    }
    __syncthreads();

    const float ca = cura[b];

    // H A-fragments, reused by phases 1 and 3
    v16h aH[4];
#pragma unroll
    for (int kt = 0; kt < 4; ++kt) aH[kt] = load_a(Hl, r, half, kt);

    // ---- phase 1: F-gate preactivation (batch 8 B-frags per K-step) ----
    v8f accF[8];
#pragma unroll
    for (int ft = 0; ft < 8; ++ft) accF[ft] = {};
#pragma unroll
    for (int kt = 0; kt < 4; ++kt) {
        v16h bw[8];
#pragma unroll
        for (int ft = 0; ft < 8; ++ft)
            bw[ft] = *(const v16h*)(pack + PK_WF + (size_t)(kt * 8 + ft) * 512 + lane * 16);
#pragma unroll
        for (int ft = 0; ft < 8; ++ft)
            accF[ft] = wmma16(aH[kt], bw[ft], accF[ft]);
    }
    // sigmoid, keep f32 in regs + stash f16 to LDS for the Uc matmul
#pragma unroll
    for (int ft = 0; ft < 8; ++ft) {
        int f = ft * 16 + r;
        float ct = ca * cols[3 * 128 + f] + cols[7 * 128 + f] + bf[f];
#pragma unroll
        for (int v = 0; v < 8; ++v) {
            float g = fast_sigmoid(accF[ft][v] + ct);
            accF[ft][v] = g;
            Gl[(v + 8 * half) * 136 + f] = (_Float16)g;
        }
    }
    __syncthreads();

    // ---- phase 2: Q = F_gate @ Uc ----
    v8f accQ[8];
#pragma unroll
    for (int ft = 0; ft < 8; ++ft) accQ[ft] = {};
#pragma unroll
    for (int kt = 0; kt < 4; ++kt) {
        v16h a = load_a(Gl, r, half, kt);
        v16h bw[8];
#pragma unroll
        for (int ft = 0; ft < 8; ++ft)
            bw[ft] = *(const v16h*)(pack + PK_UC + (size_t)(kt * 8 + ft) * 512 + lane * 16);
#pragma unroll
        for (int ft = 0; ft < 8; ++ft)
            accQ[ft] = wmma16(a, bw[ft], accQ[ft]);
    }

    // ---- phase 3: I / O / C gates per N-tile + fused epilogue ----
#pragma unroll
    for (int ft = 0; ft < 8; ++ft) {
        v8f ai = {}, ao = {}, ac = {};
#pragma unroll
        for (int kt = 0; kt < 4; ++kt) {
            size_t tb = (size_t)(kt * 8 + ft) * 512 + lane * 16;
            v16h bwi = *(const v16h*)(pack + PK_WI + tb);
            v16h bwo = *(const v16h*)(pack + PK_WO + tb);
            v16h bwc = *(const v16h*)(pack + PK_WC + tb);
            ai = wmma16(aH[kt], bwi, ai);
            ao = wmma16(aH[kt], bwo, ao);
            ac = wmma16(aH[kt], bwc, ac);
        }
        int f = ft * 16 + r;
        float ctI = ca * cols[2 * 128 + f] + cols[6 * 128 + f] + bf[f];
        float ctO = ca * cols[4 * 128 + f] + cols[8 * 128 + f] + bo[f];
        float ctC = ca * cols[5 * 128 + f] + bc[f];
#pragma unroll
        for (int v = 0; v < 8; ++v) {
            float gi = fast_sigmoid(ai[v] + ctI);
            float go = fast_sigmoid(ao[v] + ctO);
            float c_ = fast_tanh(ac[v] + accQ[ft][v] + ctC);
            float gf = accF[ft][v];
            float Ct = gf + gi * c_;
            out[(size_t)(t0 + v + 8 * half) * 128 + f] = go * fast_tanh(Ct);
        }
    }
}

extern "C" void kernel_launch(void* const* d_in, const int* in_sizes, int n_in,
                              void* d_out, int out_size, void* d_ws, size_t ws_size,
                              hipStream_t stream) {
    (void)in_sizes; (void)n_in; (void)out_size; (void)ws_size;
    const float* send  = (const float*)d_in[0];
    const float* recv  = (const float*)d_in[1];
    const float* Win   = (const float*)d_in[2];
    const float* b_in  = (const float*)d_in[3];
    const float* Wout  = (const float*)d_in[4];
    const float* b_out = (const float*)d_in[5];
    const float* Wa    = (const float*)d_in[6];
    const float* Wi    = (const float*)d_in[7];
    const float* Ui    = (const float*)d_in[8];
    const float* Wf    = (const float*)d_in[9];
    const float* Uf    = (const float*)d_in[10];
    const float* bf    = (const float*)d_in[11];
    const float* Wo    = (const float*)d_in[12];
    const float* Uo    = (const float*)d_in[13];
    const float* bo    = (const float*)d_in[14];
    const float* Wc    = (const float*)d_in[15];
    const float* Uc    = (const float*)d_in[16];
    const float* bc    = (const float*)d_in[17];

    char* w = (char*)d_ws;
    float*     SUMS = (float*)(w + WS_SUMS);
    float*     COLS = (float*)(w + WS_COLS);
    float*     CURA = (float*)(w + WS_CURA);
    _Float16*  PACK = (_Float16*)(w + WS_PACK);
    float*     PART = (float*)(w + WS_PART);
    _Float16*  Hws  = (_Float16*)(w + WS_H);
    float*     outp = (float*)d_out;

    k_rowsum<<<64, 256, 0, stream>>>(send, recv, SUMS);
    k_colsum<<<9, 128, 0, stream>>>(Win, Wout, Wi, Wf, Wo, Wc, Ui, Uf, Uo, COLS);
    k_pack<<<320, 256, 0, stream>>>(Wi, Wf, Wo, Wc, Uc, PACK);
    k_embed<<<4096, 256, 0, stream>>>(send, recv, b_in, b_out, SUMS, COLS, Hws, PART);
    k_cura<<<32, 128, 0, stream>>>(PART, Wa, CURA, outp + (size_t)BN * 128);
    k_gates<<<BN / 16, 32, 0, stream>>>(Hws, PACK, COLS, CURA, bf, bo, bc, outp);
}
